// SwinTransformerBlock_50654844289707
// MI455X (gfx1250) — matrix operand order repaired and checked by reading, not compiled, not run on hardware
//
#include <hip/hip_runtime.h>
#include <math.h>

typedef _Float16 f16;
typedef __attribute__((ext_vector_type(8)))  _Float16 v8h;
typedef __attribute__((ext_vector_type(16))) _Float16 v16h;
typedef __attribute__((ext_vector_type(8)))  float    v8f;

#define NH_    6
#define C_     192
#define HH_    112
#define WW_    112
#define N_     49
#define TOK_   200704   // 4096 windows * 49 tokens
#define DH_    768
#define SCALE_ 0.17677669529663687f  // 32^-0.5

// ---------------------------------------------------------------------------
// WMMA 16x64 per-wave tile: Out[m0:m0+16, n0:n0+64] = A[MxK] @ W[NxK]^T.
// A row-major f16, W row-major f16 (torch linear weight layout).
// One A-fragment load per k-step feeds 4 WMMAs (4x less A-side HBM traffic).
// Fragment k order: contiguous, k = 16*(lane/16) + s, same for A and W (wave32).
// ---------------------------------------------------------------------------
struct Acc4 { v8f a[4]; };

template <int K, bool UNROLL>
__device__ __forceinline__ Acc4 wmma_tile16x64(const f16* __restrict__ A,
                                               const f16* __restrict__ W,
                                               size_t m0, size_t n0) {
    const int lane = threadIdx.x & 31;
    const int hi   = lane >> 4;     // 0: lanes 0-15, 1: lanes 16-31
    const int r    = lane & 15;
    const f16* arow = A + (m0 + (size_t)r) * (size_t)K;
    const f16* wrow = W + (n0 + (size_t)r) * (size_t)K;
    Acc4 acc;
#pragma unroll
    for (int j = 0; j < 4; ++j)
        acc.a[j] = (v8f){0.f, 0.f, 0.f, 0.f, 0.f, 0.f, 0.f, 0.f};
#pragma unroll(UNROLL ? K / 32 : 2)
    for (int k0 = 0; k0 < K; k0 += 32) {
        v8h a0 = *(const v8h*)(arow + k0 + hi * 16);
        v8h a1 = *(const v8h*)(arow + k0 + hi * 16 + 8);
        v16h a = __builtin_shufflevector(a0, a1, 0,1,2,3,4,5,6,7,8,9,10,11,12,13,14,15);
#pragma unroll
        for (int j = 0; j < 4; ++j) {
            v16h b = *(const v16h*)(wrow + (size_t)j * 16 * K + k0 + hi * 16);
            acc.a[j] = __builtin_amdgcn_wmma_f32_16x16x32_f16(false, a, false, b,
                                                              (short)0, acc.a[j],
                                                              false, false);
        }
    }
    return acc;
}

// ---------------------------------------------------------------------------
// f32 -> f16 elementwise (weights)
// ---------------------------------------------------------------------------
__global__ __launch_bounds__(256) void cvt_f16_kernel(const float* __restrict__ in,
                                                      f16* __restrict__ out, int n) {
    int i = blockIdx.x * 256 + threadIdx.x;
    if (i < n) out[i] = (f16)in[i];
}

// ---------------------------------------------------------------------------
// LN1 + roll(-3,-3) + window partition.  One token per wave (8 waves / block).
// Output xw[t*192 + c] f16 in window order.
// ---------------------------------------------------------------------------
__global__ __launch_bounds__(256) void ln1_partition_kernel(const float* __restrict__ x,
                                                            const float* __restrict__ g,
                                                            const float* __restrict__ bt,
                                                            f16* __restrict__ xw) {
    const int wave = threadIdx.x >> 5;
    const int lane = threadIdx.x & 31;
    const size_t t = (size_t)blockIdx.x * 8 + wave;        // window-ordered token
    const int win = (int)(t / N_), n = (int)(t % N_);
    const int b = win >> 8, rem = win & 255, wh = rem >> 4, wc = rem & 15;
    int hs = wh * 7 + n / 7;
    int ws = wc * 7 + n % 7;
    int hi = hs + 3; if (hi >= HH_) hi -= HH_;             // roll(-3): shifted[i]=x[i+3]
    int wi = ws + 3; if (wi >= WW_) wi -= WW_;
    const float* row = x + (((size_t)b * HH_ + hi) * WW_ + wi) * C_;

    float v[6], s = 0.f, s2 = 0.f;
#pragma unroll
    for (int i = 0; i < 6; ++i) {
        v[i] = row[lane + 32 * i];
        s += v[i];
        s2 += v[i] * v[i];
    }
#pragma unroll
    for (int off = 16; off >= 1; off >>= 1) {
        s  += __shfl_xor(s,  off, 32);
        s2 += __shfl_xor(s2, off, 32);
    }
    const float mean = s * (1.f / C_);
    const float var  = s2 * (1.f / C_) - mean * mean;
    const float rs   = rsqrtf(var + 1e-5f);
    f16* orow = xw + t * C_;
#pragma unroll
    for (int i = 0; i < 6; ++i) {
        int c = lane + 32 * i;
        orow[c] = (f16)((v[i] - mean) * rs * g[c] + bt[c]);
    }
}

// ---------------------------------------------------------------------------
// LN2: same reduction over a contiguous f32 row of x2 (=d_out), output f16.
// ---------------------------------------------------------------------------
__global__ __launch_bounds__(256) void ln2_kernel(const float* __restrict__ x2,
                                                  const float* __restrict__ g,
                                                  const float* __restrict__ bt,
                                                  f16* __restrict__ xn) {
    const int wave = threadIdx.x >> 5;
    const int lane = threadIdx.x & 31;
    const size_t t = (size_t)blockIdx.x * 8 + wave;
    const float* row = x2 + t * C_;
    float v[6], s = 0.f, s2 = 0.f;
#pragma unroll
    for (int i = 0; i < 6; ++i) {
        v[i] = row[lane + 32 * i];
        s += v[i];
        s2 += v[i] * v[i];
    }
#pragma unroll
    for (int off = 16; off >= 1; off >>= 1) {
        s  += __shfl_xor(s,  off, 32);
        s2 += __shfl_xor(s2, off, 32);
    }
    const float mean = s * (1.f / C_);
    const float var  = s2 * (1.f / C_) - mean * mean;
    const float rs   = rsqrtf(var + 1e-5f);
    f16* orow = xn + t * C_;
#pragma unroll
    for (int i = 0; i < 6; ++i) {
        int c = lane + 32 * i;
        orow[c] = (f16)((v[i] - mean) * rs * g[c] + bt[c]);
    }
}

// ---------------------------------------------------------------------------
// QKV GEMM: out f16 = xw @ qkv_w^T + qkv_b.  M=200704, N=576, K=192.
// 4 waves / block, one 16x64 tile each.
// ---------------------------------------------------------------------------
__global__ __launch_bounds__(128) void qkv_gemm_kernel(const f16* __restrict__ A,
                                                       const f16* __restrict__ W,
                                                       const float* __restrict__ bias,
                                                       f16* __restrict__ out) {
    const int wv = threadIdx.x >> 5;
    const size_t m0 = ((size_t)blockIdx.y * 4 + wv) * 16;
    const size_t n0 = (size_t)blockIdx.x * 64;
    Acc4 acc = wmma_tile16x64<C_, true>(A, W, m0, n0);
    const int lane = threadIdx.x & 31, col = lane & 15, hi = lane >> 4;
#pragma unroll
    for (int j = 0; j < 4; ++j) {
        const size_t cg = n0 + j * 16 + col;
        const float bv = bias[cg];
#pragma unroll
        for (int e = 0; e < 8; ++e) {
            size_t row = m0 + e + hi * 8;
            out[row * 576 + cg] = (f16)(acc.a[j][e] + bv);
        }
    }
}

// ---------------------------------------------------------------------------
// Windowed attention: one (window, head) per block, 4 waves.
// ---------------------------------------------------------------------------
__global__ __launch_bounds__(128) void attn_kernel(const f16* __restrict__ qkv,
                                                   const float* __restrict__ mask,
                                                   const int* __restrict__ rpi,
                                                   const float* __restrict__ rpb,
                                                   f16* __restrict__ attn_out) {
    __shared__ f16   sQ[64][32];
    __shared__ f16   sK[64][32];
    __shared__ f16   sVT[32][64];   // V transposed: sVT[d][k]
    __shared__ float sS[64][64];
    __shared__ f16   sP[64][64];

    const int w = blockIdx.x, head = blockIdx.y;
    const int tid = threadIdx.x;

    // Stage 0: load q (pre-scaled), k, v tiles with zero padding 49->64.
    for (int idx = tid; idx < 64 * 32; idx += 128) {
        int r = idx >> 5, d = idx & 31;
        float qv = 0.f, kv = 0.f, vv = 0.f;
        if (r < N_) {
            size_t base = ((size_t)w * N_ + r) * 576 + head * 32 + d;
            qv = (float)qkv[base] * SCALE_;
            kv = (float)qkv[base + 192];
            vv = (float)qkv[base + 384];
        }
        sQ[r][d] = (f16)qv;
        sK[r][d] = (f16)kv;
        sVT[d][r] = (f16)vv;
    }
    __syncthreads();

    const int wv = tid >> 5, lane = tid & 31, col = lane & 15, hi = lane >> 4;
    const int m0 = wv * 16;
    const int maskbase = (w & 255) * N_ * N_;

    // Stage 1: S = (q*scale) @ k^T + bias + mask   (K = 32, one WMMA per tile)
    {
        v8h a0 = *(const v8h*)(&sQ[m0 + col][hi * 16]);
        v8h a1 = *(const v8h*)(&sQ[m0 + col][hi * 16 + 8]);
        v16h a = __builtin_shufflevector(a0, a1, 0,1,2,3,4,5,6,7,8,9,10,11,12,13,14,15);
#pragma unroll
        for (int j = 0; j < 4; ++j) {
            const int n0 = j * 16;
            v16h b = *(const v16h*)(&sK[n0 + col][hi * 16]);
            v8f acc = {0.f, 0.f, 0.f, 0.f, 0.f, 0.f, 0.f, 0.f};
            acc = __builtin_amdgcn_wmma_f32_16x16x32_f16(false, a, false, b,
                                                         (short)0, acc, false, false);
#pragma unroll
            for (int e = 0; e < 8; ++e) {
                int row = m0 + e + hi * 8;
                int cc = n0 + col;
                float val;
                if (row < N_ && cc < N_) {
                    int lin = row * N_ + cc;
                    val = acc[e] + rpb[rpi[lin] * NH_ + head] + mask[maskbase + lin];
                } else {
                    val = -1e30f;
                }
                sS[row][cc] = val;
            }
        }
    }
    __syncthreads();

    // Stage 2: row softmax (valid cols 0..48), write P f16 with zero padding.
    if (tid < 64) {
        const int row = tid;
        if (row < N_) {
            float mx = -1e30f;
            for (int c = 0; c < N_; ++c) mx = fmaxf(mx, sS[row][c]);
            float sum = 0.f;
            for (int c = 0; c < N_; ++c) {
                float e = __expf(sS[row][c] - mx);
                sum += e;
                sP[row][c] = (f16)e;
            }
            const float inv = 1.f / sum;
            for (int c = 0; c < N_; ++c) sP[row][c] = (f16)((float)sP[row][c] * inv);
            for (int c = N_; c < 64; ++c) sP[row][c] = (f16)0.f;
        } else {
            for (int c = 0; c < 64; ++c) sP[row][c] = (f16)0.f;
        }
    }
    __syncthreads();

    // Stage 3: out = P @ V  (K = 64 -> two WMMA k-steps, two 16-col tiles)
#pragma unroll
    for (int nt = 0; nt < 2; ++nt) {
        const int n0 = nt * 16;
        v8f acc = {0.f, 0.f, 0.f, 0.f, 0.f, 0.f, 0.f, 0.f};
#pragma unroll
        for (int ks = 0; ks < 2; ++ks) {
            const int k0 = ks * 32;
            v8h a0 = *(const v8h*)(&sP[m0 + col][k0 + hi * 16]);
            v8h a1 = *(const v8h*)(&sP[m0 + col][k0 + hi * 16 + 8]);
            v16h a = __builtin_shufflevector(a0, a1, 0,1,2,3,4,5,6,7,8,9,10,11,12,13,14,15);
            v16h b = *(const v16h*)(&sVT[n0 + col][k0 + hi * 16]);
            acc = __builtin_amdgcn_wmma_f32_16x16x32_f16(false, a, false, b,
                                                         (short)0, acc, false, false);
        }
#pragma unroll
        for (int e = 0; e < 8; ++e) {
            int row = m0 + e + hi * 8;
            if (row < N_)
                attn_out[((size_t)w * N_ + row) * C_ + head * 32 + n0 + col] = (f16)acc[e];
        }
    }
}

// ---------------------------------------------------------------------------
// Proj GEMM + window reverse + roll(+3,+3) + residual -> x2 (f32, in d_out).
// ---------------------------------------------------------------------------
__global__ __launch_bounds__(128) void proj_gemm_kernel(const f16* __restrict__ A,
                                                        const f16* __restrict__ W,
                                                        const float* __restrict__ bias,
                                                        const float* __restrict__ xin,
                                                        float* __restrict__ x2) {
    const int wv = threadIdx.x >> 5;
    const size_t m0 = ((size_t)blockIdx.y * 4 + wv) * 16;
    const size_t n0 = (size_t)blockIdx.x * 64;
    Acc4 acc = wmma_tile16x64<C_, true>(A, W, m0, n0);
    const int lane = threadIdx.x & 31, col = lane & 15, hi = lane >> 4;
#pragma unroll
    for (int e = 0; e < 8; ++e) {
        size_t t = m0 + e + hi * 8;
        int win = (int)(t / N_), n = (int)(t % N_);
        int b = win >> 8, rem = win & 255, wh = rem >> 4, wc = rem & 15;
        int hs = wh * 7 + n / 7;
        int ws = wc * 7 + n % 7;
        int himg = hs + 3; if (himg >= HH_) himg -= HH_;   // roll(+3): out[hs+3]=shifted[hs]
        int wimg = ws + 3; if (wimg >= WW_) wimg -= WW_;
        size_t obase = (((size_t)b * HH_ + himg) * WW_ + wimg) * C_;
#pragma unroll
        for (int j = 0; j < 4; ++j) {
            size_t cg = n0 + j * 16 + col;
            x2[obase + cg] = xin[obase + cg] + acc.a[j][e] + bias[cg];
        }
    }
}

// ---------------------------------------------------------------------------
// FC1 GEMM + exact GELU -> h f16.  M=200704, N=768, K=192.
// ---------------------------------------------------------------------------
__global__ __launch_bounds__(128) void fc1_gemm_kernel(const f16* __restrict__ A,
                                                       const f16* __restrict__ W,
                                                       const float* __restrict__ bias,
                                                       f16* __restrict__ out) {
    const int wv = threadIdx.x >> 5;
    const size_t m0 = ((size_t)blockIdx.y * 4 + wv) * 16;
    const size_t n0 = (size_t)blockIdx.x * 64;
    Acc4 acc = wmma_tile16x64<C_, true>(A, W, m0, n0);
    const int lane = threadIdx.x & 31, col = lane & 15, hi = lane >> 4;
#pragma unroll
    for (int j = 0; j < 4; ++j) {
        const size_t cg = n0 + j * 16 + col;
        const float bv = bias[cg];
#pragma unroll
        for (int e = 0; e < 8; ++e) {
            size_t row = m0 + e + hi * 8;
            float v = acc.a[j][e] + bv;
            v = 0.5f * v * (1.f + erff(v * 0.70710678118654752f));
            out[row * DH_ + cg] = (f16)v;
        }
    }
}

// ---------------------------------------------------------------------------
// FC2 GEMM + residual (in place on d_out).  M=200704, N=192, K=768.
// ---------------------------------------------------------------------------
__global__ __launch_bounds__(128) void fc2_gemm_kernel(const f16* __restrict__ A,
                                                       const f16* __restrict__ W,
                                                       const float* __restrict__ bias,
                                                       float* __restrict__ xio) {
    const int wv = threadIdx.x >> 5;
    const size_t m0 = ((size_t)blockIdx.y * 4 + wv) * 16;
    const size_t n0 = (size_t)blockIdx.x * 64;
    Acc4 acc = wmma_tile16x64<DH_, false>(A, W, m0, n0);
    const int lane = threadIdx.x & 31, col = lane & 15, hi = lane >> 4;
#pragma unroll
    for (int j = 0; j < 4; ++j) {
        const size_t cg = n0 + j * 16 + col;
        const float bv = bias[cg];
#pragma unroll
        for (int e = 0; e < 8; ++e) {
            size_t idx = (m0 + e + hi * 8) * C_ + cg;
            xio[idx] = xio[idx] + acc.a[j][e] + bv;
        }
    }
}

// ---------------------------------------------------------------------------
extern "C" void kernel_launch(void* const* d_in, const int* in_sizes, int n_in,
                              void* d_out, int out_size, void* d_ws, size_t ws_size,
                              hipStream_t stream) {
    const float* x      = (const float*)d_in[0];
    const float* mask   = (const float*)d_in[1];
    const int*   rpi    = (const int*)d_in[2];
    const float* rpb    = (const float*)d_in[3];
    const float* n1g    = (const float*)d_in[4];
    const float* n1b    = (const float*)d_in[5];
    const float* qkv_w  = (const float*)d_in[6];
    const float* qkv_b  = (const float*)d_in[7];
    const float* proj_w = (const float*)d_in[8];
    const float* proj_b = (const float*)d_in[9];
    const float* n2g    = (const float*)d_in[10];
    const float* n2b    = (const float*)d_in[11];
    const float* fc1_w  = (const float*)d_in[12];
    const float* fc1_b  = (const float*)d_in[13];
    const float* fc2_w  = (const float*)d_in[14];
    const float* fc2_b  = (const float*)d_in[15];
    float* out = (float*)d_out;

    char* ws = (char*)d_ws;
    size_t off = 0;
    auto carve = [&](size_t bytes) {
        char* p = ws + off;
        off += (bytes + 255) & ~(size_t)255;
        return p;
    };
    // region1: xw (f16) -> attn_out (f16) -> xn2 (f16), all 200704x192
    f16* region1 = (f16*)carve((size_t)TOK_ * C_ * 2);
    // region2: qkv (200704x576 f16) then h (200704x768 f16)
    f16* region2 = (f16*)carve((size_t)TOK_ * DH_ * 2);
    // f16 weights
    f16* qkvw_h = (f16*)carve((size_t)442368 * 2);
    f16* projw_h = qkvw_h + 110592;   // byte offsets all 32B-aligned
    f16* fc1w_h  = qkvw_h + 147456;
    f16* fc2w_h  = qkvw_h + 294912;

    // 1. weight conversion
    cvt_f16_kernel<<<(110592 + 255) / 256, 256, 0, stream>>>(qkv_w, qkvw_h, 110592);
    cvt_f16_kernel<<<(36864 + 255) / 256, 256, 0, stream>>>(proj_w, projw_h, 36864);
    cvt_f16_kernel<<<(147456 + 255) / 256, 256, 0, stream>>>(fc1_w, fc1w_h, 147456);
    cvt_f16_kernel<<<(147456 + 255) / 256, 256, 0, stream>>>(fc2_w, fc2w_h, 147456);

    // 2. LN1 + shift + window partition
    ln1_partition_kernel<<<TOK_ / 8, 256, 0, stream>>>(x, n1g, n1b, region1);

    // 3. QKV GEMM (M=200704, N=576, K=192), 16x64 tiles
    qkv_gemm_kernel<<<dim3(576 / 64, TOK_ / 64), 128, 0, stream>>>(region1, qkvw_h, qkv_b, region2);

    // 4. attention (4096 windows x 6 heads) -> region1
    attn_kernel<<<dim3(4096, NH_), 128, 0, stream>>>(region2, mask, rpi, rpb, region1);

    // 5. proj + reverse + roll + residual -> d_out (x2)
    proj_gemm_kernel<<<dim3(C_ / 64, TOK_ / 64), 128, 0, stream>>>(region1, projw_h, proj_b, x, out);

    // 6. LN2 -> region1 (xn2 f16)
    ln2_kernel<<<TOK_ / 8, 256, 0, stream>>>(out, n2g, n2b, region1);

    // 7. FC1 + GELU -> region2 (h f16)
    fc1_gemm_kernel<<<dim3(DH_ / 64, TOK_ / 64), 128, 0, stream>>>(region1, fc1w_h, fc1_b, region2);

    // 8. FC2 + residual (in place on d_out)
    fc2_gemm_kernel<<<dim3(C_ / 64, TOK_ / 64), 128, 0, stream>>>(region2, fc2w_h, fc2_b, out);
}
